// DynamicRetrievalAttention_23038204575834
// MI455X (gfx1250) — compile-verified, matrix operand order checked by reading, not run on hardware
//
#include <hip/hip_runtime.h>
#include <hip/hip_bf16.h>

#define NUM_HEADS  8
#define NUM_FRAMES 40
#define FRAME_HW   780
#define TOP_K      10
#define WINDOW     5
#define DIM        512
#define HEAD_DIM   64
#define STOT       (WINDOW + TOP_K)                // 15 selected frames
#define KBLK       32
#define NKBLK      ((FRAME_HW + KBLK - 1) / KBLK)  // 25
#define TAILCNT    (FRAME_HW - (NKBLK - 1) * KBLK) // 12 valid keys in tail block
#define KSTRIDE    72                              // K_lds row stride (bf16), 144B rows
#define VSTRIDE    40                              // V_lds row stride (bf16), 80B rows
#define QT         2                               // query 16-row tiles per wave
#define SCALE_LOG2E 0.18033688011112f              // (1/sqrt(64)) * log2(e), folded into Q

#define SEQ_ELEMS  ((size_t)NUM_FRAMES * FRAME_HW * DIM)   // 15,974,400
#define WS_SEL_BYTES 4096                                   // sel region (aligned)

typedef __attribute__((ext_vector_type(16))) __bf16 v16bf;
typedef __attribute__((ext_vector_type(8)))  __bf16 v8bf;
typedef __attribute__((ext_vector_type(4)))  __bf16 v4bf;
typedef __attribute__((ext_vector_type(8)))  float  v8f;
typedef __attribute__((ext_vector_type(4)))  float  v4f;

union ABPack { v16bf v; v8bf h[2]; __bf16 e[16]; };

__device__ __forceinline__ float fast_exp2(float x) {
#if __has_builtin(__builtin_amdgcn_exp2f)
    return __builtin_amdgcn_exp2f(x);          // raw v_exp_f32 (base-2)
#else
    return exp2f(x);
#endif
}

// Build a 16x32 A-operand / 32x16 B-operand register tile from two aligned
// 16B LDS chunks (even/odd K-chunk lane-half interleave handled by caller).
__device__ __forceinline__ v16bf ld_ab(const __bf16* p0, const __bf16* p1) {
    ABPack u;
    u.h[0] = *(const v8bf*)p0;
    u.h[1] = *(const v8bf*)p1;
    return u.v;
}

// ---- staging, f32 source (fallback path): convert while copying ----
__device__ __forceinline__ void stage_kv_f32(const float* __restrict__ kfp,
                                             const float* __restrict__ vfp,
                                             int kbase, __bf16* Kb, __bf16* Vb, int tid) {
    {   // K row-major: thread = (key row, 16-col slab)
        const int srow = tid >> 2;
        const int scol = (tid & 3) * 16;
        int kpos = kbase + srow; if (kpos >= FRAME_HW) kpos = FRAME_HW - 1;
        const float* krp = kfp + (size_t)kpos * DIM + scol;
        v8bf b0, b1;
        #pragma unroll
        for (int i = 0; i < 8; ++i) { b0[i] = (__bf16)krp[i]; b1[i] = (__bf16)krp[8 + i]; }
        *(v8bf*)&Kb[srow * KSTRIDE + scol]     = b0;
        *(v8bf*)&Kb[srow * KSTRIDE + scol + 8] = b1;
    }
    {   // V^T: thread = 4 keys x 4 hd; coalesced loads, packed b64 stores
        const int h0 = (tid & 15) * 4;
        const int k0 = (tid >> 4) * 4;
        float val[4][4];
        #pragma unroll
        for (int j = 0; j < 4; ++j) {
            int kp = kbase + k0 + j; if (kp >= FRAME_HW) kp = FRAME_HW - 1;
            const float* vrp = vfp + (size_t)kp * DIM + h0;
            #pragma unroll
            for (int i = 0; i < 4; ++i) val[j][i] = vrp[i];
        }
        #pragma unroll
        for (int i = 0; i < 4; ++i) {
            v4bf p;
            #pragma unroll
            for (int j = 0; j < 4; ++j) p[j] = (__bf16)val[j][i];
            *(v4bf*)&Vb[(h0 + i) * VSTRIDE + k0] = p;
        }
    }
}

// ---- staging, pre-converted bf16 source: pure copy, zero conversion VALU ----
__device__ __forceinline__ void stage_kv_bf16(const __bf16* __restrict__ kfp,
                                              const __bf16* __restrict__ vfp,
                                              int kbase, __bf16* Kb, __bf16* Vb, int tid) {
    {   // K row-major: 2x b128 load -> 2x b128 ds store
        const int srow = tid >> 2;
        const int scol = (tid & 3) * 16;
        int kpos = kbase + srow; if (kpos >= FRAME_HW) kpos = FRAME_HW - 1;
        const __bf16* krp = kfp + (size_t)kpos * DIM + scol;
        v8bf b0 = *(const v8bf*)krp;
        v8bf b1 = *(const v8bf*)(krp + 8);
        *(v8bf*)&Kb[srow * KSTRIDE + scol]     = b0;
        *(v8bf*)&Kb[srow * KSTRIDE + scol + 8] = b1;
    }
    {   // V^T: 4x b64 loads, register 4x4 transpose, 4x b64 ds stores
        const int h0 = (tid & 15) * 4;
        const int k0 = (tid >> 4) * 4;
        v4bf row[4];
        #pragma unroll
        for (int j = 0; j < 4; ++j) {
            int kp = kbase + k0 + j; if (kp >= FRAME_HW) kp = FRAME_HW - 1;
            row[j] = *(const v4bf*)(vfp + (size_t)kp * DIM + h0);
        }
        #pragma unroll
        for (int i = 0; i < 4; ++i) {
            v4bf p;
            #pragma unroll
            for (int j = 0; j < 4; ++j) p[j] = row[j][i];
            *(v4bf*)&Vb[(h0 + i) * VSTRIDE + k0] = p;
        }
    }
}

// ---------------------------------------------------------------------------
// Kernel 0: masked top-k frame selection.  sel[f] = [5 local window, 10 top-k]
// ---------------------------------------------------------------------------
__global__ void dra_topk_kernel(const float* __restrict__ sim, int* __restrict__ sel) {
    const int f = threadIdx.x;
    if (f >= NUM_FRAMES) return;
    float vals[NUM_FRAMES];
    for (int i = 0; i < NUM_FRAMES; ++i) vals[i] = sim[f * NUM_FRAMES + i];
    int start = f - WINDOW / 2;
    if (start < 0) start = 0;
    if (start > NUM_FRAMES - WINDOW) start = NUM_FRAMES - WINDOW;
    for (int w = 0; w < WINDOW; ++w) {
        sel[f * STOT + w] = start + w;
        vals[start + w] = -1e9f;
    }
    for (int t = 0; t < TOP_K; ++t) {
        int bi = 0; float bv = vals[0];
        for (int i = 1; i < NUM_FRAMES; ++i)
            if (vals[i] > bv) { bv = vals[i]; bi = i; }
        sel[f * STOT + WINDOW + t] = bi;
        vals[bi] = -1e30f;
    }
}

// ---------------------------------------------------------------------------
// Kernel 1: one-shot f32 -> bf16 conversion of K/V into workspace.
// ---------------------------------------------------------------------------
__global__ __launch_bounds__(256)
void dra_cvt_kernel(const float* __restrict__ src, __bf16* __restrict__ dst, int n4) {
    const int i = blockIdx.x * blockDim.x + threadIdx.x;
    if (i >= n4) return;
    v4f x = *(const v4f*)(src + (size_t)i * 4);
    v4bf y;
    #pragma unroll
    for (int j = 0; j < 4; ++j) y[j] = (__bf16)x[j];
    *(v4bf*)(dst + (size_t)i * 4) = y;
}

// ---------------------------------------------------------------------------
// Kernel 2: double-buffered bf16-WMMA flash attention over 15 selected frames.
// PRE=true: K/V staged from pre-converted bf16 (pure copies in hot loop).
// ---------------------------------------------------------------------------
template <bool PRE>
__global__ __launch_bounds__(128)
void dra_attn_kernel(const float* __restrict__ q,
                     const float* __restrict__ kf, const float* __restrict__ vf,
                     const __bf16* __restrict__ kb, const __bf16* __restrict__ vb,
                     const int* __restrict__ sel, float* __restrict__ out) {
    __shared__ __align__(16) __bf16 Klds[2][KBLK * KSTRIDE];
    __shared__ __align__(16) __bf16 Vlds[2][HEAD_DIM * VSTRIDE];

    const int tid  = threadIdx.x;
    const int lane = tid & 31;
    const int wid  = tid >> 5;
    const int lh   = lane >> 4;
    const int l16  = lane & 15;

    const int f  = blockIdx.y;
    const int h  = blockIdx.z;
    const int qb = blockIdx.x * (4 * QT * 16) + wid * (QT * 16);

    // ---- Q as B-operand (32hd x 16q), prescaled by scale*log2(e) ----
    v16bf bq[QT][2];
    int   qvalid[QT];
    #pragma unroll
    for (int qt = 0; qt < QT; ++qt) {
        int qrow = qb + qt * 16 + l16;
        qvalid[qt] = (qrow < FRAME_HW);
        if (qrow >= FRAME_HW) qrow = FRAME_HW - 1;      // clamp: no divergence
        const float* qp = q + ((size_t)(f * FRAME_HW + qrow)) * DIM + h * HEAD_DIM;
        #pragma unroll
        for (int c = 0; c < 2; ++c) {
            ABPack u;
            const int base = c * 32 + lh * 8;
            #pragma unroll
            for (int i = 0; i < 8; ++i) {
                u.e[i]     = (__bf16)(qp[base + i]      * SCALE_LOG2E);
                u.e[8 + i] = (__bf16)(qp[base + 16 + i] * SCALE_LOG2E);
            }
            bq[qt][c] = u.v;
        }
    }

    v8f   acc[QT][4] = {};
    float mstate[QT], lstate[QT];
    #pragma unroll
    for (int qt = 0; qt < QT; ++qt) { mstate[qt] = -1e30f; lstate[qt] = 0.0f; }

    const int* selp = sel + f * STOT;
    const int TOTAL = STOT * NKBLK;                     // 375

    const size_t hoff = (size_t)h * HEAD_DIM;

    // prologue: stage block 0 into buffer 0
    {
        const int fr0 = selp[0];
        const size_t base = (size_t)fr0 * FRAME_HW * DIM + hoff;
        if (PRE) stage_kv_bf16(kb + base, vb + base, 0, Klds[0], Vlds[0], tid);
        else     stage_kv_f32 (kf + base, vf + base, 0, Klds[0], Vlds[0], tid);
    }

    for (int it = 0; it < TOTAL; ++it) {
        __syncthreads();                                // buf[it&1] staged & free
        const int buf = it & 1;

        // ---- stage next block into the other buffer (hides global latency) ----
        if (it + 1 < TOTAL) {
            const int nit = it + 1;
            const int nsi = nit / NKBLK;
            const int nbl = nit - nsi * NKBLK;
            const int frn = selp[nsi];                  // uniform scalar load
            const size_t base = (size_t)frn * FRAME_HW * DIM + hoff;
            if (PRE) stage_kv_bf16(kb + base, vb + base, nbl * KBLK,
                                   Klds[buf ^ 1], Vlds[buf ^ 1], tid);
            else     stage_kv_f32 (kf + base, vf + base, nbl * KBLK,
                                   Klds[buf ^ 1], Vlds[buf ^ 1], tid);
        }

        const int si   = it / NKBLK;
        const bool tail = (it - si * NKBLK) == (NKBLK - 1);
        const __bf16* Kb = Klds[buf];
        const __bf16* Vb = Vlds[buf];

        // ---- S^T = K(16x64) * Q^T(64x16): 2 key groups x 2 hd chunks ----
        v8f ct[QT][2] = {};
        #pragma unroll
        for (int g = 0; g < 2; ++g) {
            #pragma unroll
            for (int c = 0; c < 2; ++c) {
                const int key = g * 16 + l16;
                const int hb  = c * 32 + lh * 8;
                v16bf ak = ld_ab(&Kb[key * KSTRIDE + hb], &Kb[key * KSTRIDE + hb + 16]);
                #pragma unroll
                for (int qt = 0; qt < QT; ++qt)
                    ct[qt][g] = __builtin_amdgcn_wmma_f32_16x16x32_bf16(
                        false, ak, false, bq[qt][c], (short)0, ct[qt][g], false, false);
            }
        }

        // ---- online softmax (log2 domain); P packs straight into B layout ----
        v16bf bp[QT];
        float alpha[QT];
        #pragma unroll
        for (int qt = 0; qt < QT; ++qt) {
            float s0[8], s1[8];
            #pragma unroll
            for (int m = 0; m < 8; ++m) { s0[m] = ct[qt][0][m]; s1[m] = ct[qt][1][m]; }
            if (tail) {                                  // only 1 of 25 blocks
                #pragma unroll
                for (int m = 0; m < 8; ++m) {
                    if (lh * 8 + m >= TAILCNT) s0[m] = -1e30f;
                    s1[m] = -1e30f;                      // keys 16..31 all invalid
                }
            }
            float rm = -1e30f;
            #pragma unroll
            for (int m = 0; m < 8; ++m) rm = fmaxf(rm, fmaxf(s0[m], s1[m]));
            rm = fmaxf(rm, __shfl_xor(rm, 16, 32));
            const float mnew = fmaxf(mstate[qt], rm);
            alpha[qt] = fast_exp2(mstate[qt] - mnew);
            float rs = 0.0f;
            ABPack u;
            #pragma unroll
            for (int m = 0; m < 8; ++m) {
                const float p0 = fast_exp2(s0[m] - mnew);
                const float p1 = fast_exp2(s1[m] - mnew);
                rs += p0 + p1;
                u.e[m]     = (__bf16)p0;
                u.e[8 + m] = (__bf16)p1;
            }
            bp[qt] = u.v;
            rs += __shfl_xor(rs, 16, 32);
            lstate[qt] = lstate[qt] * alpha[qt] + rs;
            mstate[qt] = mnew;
        }

        // ---- O^T += V^T(16hd x 32key) * P^T(32key x 16q), 4 hd tiles ----
        #pragma unroll
        for (int t = 0; t < 4; ++t) {
            const int hd = t * 16 + l16;
            const int kbo = lh * 8;
            v16bf av = ld_ab(&Vb[hd * VSTRIDE + kbo], &Vb[hd * VSTRIDE + kbo + 16]);
            #pragma unroll
            for (int qt = 0; qt < QT; ++qt) {
                #pragma unroll
                for (int m = 0; m < 8; ++m) acc[qt][t][m] *= alpha[qt];
                acc[qt][t] = __builtin_amdgcn_wmma_f32_16x16x32_bf16(
                    false, av, false, bp[qt], (short)0, acc[qt][t], false, false);
            }
        }
    }

    // ---- write O: lane = qrow, vgprs = contiguous hd -> 16B stores ----
    #pragma unroll
    for (int qt = 0; qt < QT; ++qt) {
        if (!qvalid[qt]) continue;
        const float inv = 1.0f / lstate[qt];
        float* op = out + ((size_t)(f * FRAME_HW + qb + qt * 16 + l16)) * DIM + h * HEAD_DIM;
        #pragma unroll
        for (int t = 0; t < 4; ++t) {
            const int hb = t * 16 + lh * 8;
            v4f lo = { acc[qt][t][0] * inv, acc[qt][t][1] * inv,
                       acc[qt][t][2] * inv, acc[qt][t][3] * inv };
            v4f hi = { acc[qt][t][4] * inv, acc[qt][t][5] * inv,
                       acc[qt][t][6] * inv, acc[qt][t][7] * inv };
            *(v4f*)(op + hb)     = lo;
            *(v4f*)(op + hb + 4) = hi;
        }
    }
}

extern "C" void kernel_launch(void* const* d_in, const int* in_sizes, int n_in,
                              void* d_out, int out_size, void* d_ws, size_t ws_size,
                              hipStream_t stream) {
    const float* q   = (const float*)d_in[0];
    const float* k   = (const float*)d_in[1];
    const float* v   = (const float*)d_in[2];
    const float* sim = (const float*)d_in[3];
    float* out = (float*)d_out;

    int*    sel = (int*)d_ws;                                   // 40*15 ints
    __bf16* kb  = (__bf16*)((char*)d_ws + WS_SEL_BYTES);
    __bf16* vb  = kb + SEQ_ELEMS;
    const size_t need = WS_SEL_BYTES + 2 * SEQ_ELEMS * sizeof(__bf16);
    const bool pre = (ws_size >= need);                         // deterministic

    dra_topk_kernel<<<1, 64, 0, stream>>>(sim, sel);

    dim3 grid((FRAME_HW + (4 * QT * 16) - 1) / (4 * QT * 16), NUM_FRAMES, NUM_HEADS);
    if (pre) {
        const int n4 = (int)(SEQ_ELEMS / 4);
        dra_cvt_kernel<<<(n4 + 255) / 256, 256, 0, stream>>>(k, kb, n4);
        dra_cvt_kernel<<<(n4 + 255) / 256, 256, 0, stream>>>(v, vb, n4);
        dra_attn_kernel<true><<<grid, 128, 0, stream>>>(q, k, v, kb, vb, sel, out);
    } else {
        dra_attn_kernel<false><<<grid, 128, 0, stream>>>(q, k, v, kb, vb, sel, out);
    }
}
